// MinimalTransformerBlock_87849261072488
// MI455X (gfx1250) — compile-verified, matrix operand order checked by reading, not run on hardware
//
#include <hip/hip_runtime.h>
#include <hip/hip_bf16.h>

// ---------------------------------------------------------------------------
// Types for CDNA5 WMMA (wave32): v_wmma_f32_16x16x32_bf16
// ---------------------------------------------------------------------------
typedef __bf16 bf16x16 __attribute__((ext_vector_type(16)));
typedef __bf16 bf16x8  __attribute__((ext_vector_type(8)));
typedef float  f32x8   __attribute__((ext_vector_type(8)));

__device__ __forceinline__ f32x8 wmma_bf16(bf16x16 a, bf16x16 b, f32x8 c) {
  return __builtin_amdgcn_wmma_f32_16x16x32_bf16(
      /*neg_a=*/false, a, /*neg_b=*/false, b,
      /*c_mod=*/(short)0, c, /*reuse_a=*/false, /*reuse_b=*/false);
}

// Load one 16x32 bf16 fragment in CDNA5 A/B layout from a row-major matrix
// with leading dimension ld (elements). Works for both A (lane&15 = M row)
// and B (lane&15 = N column, source stored [N,K] row-major).
// Lanes 0-15: K = {k0..k0+7, k0+16..k0+23}; lanes 16-31: K = {+8..+15, +24..+31}.
__device__ __forceinline__ bf16x16 load_frag(const __bf16* X, int ld, int row0,
                                             int k0, int lane) {
  const int r  = row0 + (lane & 15);
  const int kk = k0 + ((lane & 16) ? 8 : 0);
  const __bf16* p = X + (size_t)r * (size_t)ld + kk;
  bf16x8 lo = *reinterpret_cast<const bf16x8*>(p);       // 16B contiguous
  bf16x8 hi = *reinterpret_cast<const bf16x8*>(p + 16);  // 16B contiguous
  bf16x16 out;
#pragma unroll
  for (int i = 0; i < 8; ++i) { out[i] = lo[i]; out[i + 8] = hi[i]; }
  return out;
}

// gfx1250 async global -> LDS copy (128 bits per lane, tracked by ASYNCcnt)
__device__ __forceinline__ void async_copy_b128(unsigned lds_addr,
                                                const void* gptr) {
  asm volatile("global_load_async_to_lds_b128 %0, %1, off"
               :: "v"(lds_addr), "v"(gptr)
               : "memory");
}
__device__ __forceinline__ void wait_async0() {
  asm volatile("s_wait_asynccnt 0" ::: "memory");
}

// ---------------------------------------------------------------------------
// fp32 -> bf16 convert (weights)
// ---------------------------------------------------------------------------
__global__ __launch_bounds__(256) void f32_to_bf16_kernel(
    const float* __restrict__ in, __bf16* __restrict__ out, int n) {
  int i = blockIdx.x * 256 + threadIdx.x;
  if (i < n) out[i] = (__bf16)in[i];
}

// ---------------------------------------------------------------------------
// LayerNorm: one 256-thread block per row of C=1024, bf16 output
// ---------------------------------------------------------------------------
__global__ __launch_bounds__(256) void ln_bf16_kernel(
    const float* __restrict__ X, const float* __restrict__ g,
    const float* __restrict__ bta, __bf16* __restrict__ out, int C) {
  const int row = blockIdx.x;
  const int tid = threadIdx.x;
  const float* x = X + (size_t)row * C;
  __shared__ float red[256];

  float s = 0.f;
  for (int i = tid; i < C; i += 256) s += x[i];
  red[tid] = s; __syncthreads();
  for (int st = 128; st > 0; st >>= 1) {
    if (tid < st) red[tid] += red[tid + st];
    __syncthreads();
  }
  const float mu = red[0] / (float)C;
  __syncthreads();

  float v = 0.f;
  for (int i = tid; i < C; i += 256) { float d = x[i] - mu; v += d * d; }
  red[tid] = v; __syncthreads();
  for (int st = 128; st > 0; st >>= 1) {
    if (tid < st) red[tid] += red[tid + st];
    __syncthreads();
  }
  const float rstd = rsqrtf(red[0] / (float)C + 1e-5f);

  __bf16* o = out + (size_t)row * C;
  for (int i = tid; i < C; i += 256)
    o[i] = (__bf16)((x[i] - mu) * rstd * g[i] + bta[i]);
}

// ---------------------------------------------------------------------------
// bf16 WMMA GEMM: C[M,N] = A[M,K] @ Bw[N,K]^T + bias (+ epilogue)
// 256 threads = 8 waves in a 2(M) x 4(N) grid; wave tile 64x32; WG tile
// 128x128. A-tile (128x64, 16KB) is double-buffer staged in LDS via async
// global->LDS copies (ASYNCcnt) -> 16 WMMAs per workgroup rendezvous;
// B fragments come straight from global/L2 (with one-chunk-ahead prefetch).
// M,N multiples of 128; K multiple of 64.
// MODE 0: f32 out + bias
// MODE 2: f32 out + bias + residual
// MODE 3: bf16 out + bias + exact GELU
// ---------------------------------------------------------------------------
template <int MODE>
__global__ __launch_bounds__(256) void gemm_bf16_kernel(
    const __bf16* __restrict__ A, const __bf16* __restrict__ Bw,
    const float* __restrict__ bias, const float* __restrict__ Rsd,
    float* __restrict__ Cf, __bf16* __restrict__ Cb, int M, int N, int K) {
  const int tid = threadIdx.x;
  const int lane = tid & 31;
  const int wave = tid >> 5;
  const int wm = wave >> 2;          // 0..1  (M)
  const int wn = wave & 3;           // 0..3  (N)
  const int mblk = blockIdx.y * 128;
  const int m0 = mblk + wm * 64;
  const int n0 = blockIdx.x * 128 + wn * 32;

  __shared__ alignas(16) __bf16 lds_a[2][128 * 64];  // 2 x 16KB double buffer

  // Async stage of one 128x64 A tile: 256 threads x 64B = 16KB
  const int st_off = tid * 64;                 // byte offset within tile
  const int st_row = st_off >> 7;              // row (128B per row)
  const int st_cb  = st_off & 127;             // byte within row (0 or 64)
  const unsigned lb0 = (unsigned)(uintptr_t)(&lds_a[0][0]);
  const unsigned lb1 = (unsigned)(uintptr_t)(&lds_a[1][0]);
  auto stage_a = [&](unsigned lbase, int k0) {
    const __bf16* g = A + (size_t)(mblk + st_row) * K + k0 + (st_cb >> 1);
    async_copy_b128(lbase + st_off, g);
    async_copy_b128(lbase + st_off + 16, (const char*)g + 16);
    async_copy_b128(lbase + st_off + 32, (const char*)g + 32);
    async_copy_b128(lbase + st_off + 48, (const char*)g + 48);
  };

  const f32x8 zero = {0.f, 0.f, 0.f, 0.f, 0.f, 0.f, 0.f, 0.f};
  f32x8 acc[4][2];
#pragma unroll
  for (int i = 0; i < 4; ++i)
#pragma unroll
    for (int j = 0; j < 2; ++j) acc[i][j] = zero;

  stage_a(lb0, 0);
  int buf = 0;
  for (int k0 = 0; k0 < K; k0 += 64) {
    wait_async0();          // our slice of current buffer landed in LDS
    __syncthreads();        // everyone's slice landed; prev reads all done
    if (k0 + 64 < K) {
      stage_a(buf ? lb0 : lb1, k0 + 64);
      // prefetch next-chunk B through L2 (global_prefetch)
      __builtin_prefetch(Bw + (size_t)(n0 + (lane & 15)) * K + k0 + 64, 0, 1);
    }
    const __bf16* la = &lds_a[buf][0];
#pragma unroll
    for (int kk = 0; kk < 64; kk += 32) {
      bf16x16 a0 = load_frag(la, 64, wm * 64, kk, lane);       // ds_load_b128
      bf16x16 a1 = load_frag(la, 64, wm * 64 + 16, kk, lane);
      bf16x16 a2 = load_frag(la, 64, wm * 64 + 32, kk, lane);
      bf16x16 a3 = load_frag(la, 64, wm * 64 + 48, kk, lane);
      bf16x16 b0 = load_frag(Bw, K, n0, k0 + kk, lane);
      bf16x16 b1 = load_frag(Bw, K, n0 + 16, k0 + kk, lane);
      acc[0][0] = wmma_bf16(a0, b0, acc[0][0]);
      acc[0][1] = wmma_bf16(a0, b1, acc[0][1]);
      acc[1][0] = wmma_bf16(a1, b0, acc[1][0]);
      acc[1][1] = wmma_bf16(a1, b1, acc[1][1]);
      acc[2][0] = wmma_bf16(a2, b0, acc[2][0]);
      acc[2][1] = wmma_bf16(a2, b1, acc[2][1]);
      acc[3][0] = wmma_bf16(a3, b0, acc[3][0]);
      acc[3][1] = wmma_bf16(a3, b1, acc[3][1]);
    }
    buf ^= 1;
  }

  // C/D layout: VGPR r -> row (r + (lane>=16 ? 8 : 0)), column = lane&15
  const int col_l = lane & 15;
  const int half8 = (lane & 16) ? 8 : 0;
#pragma unroll
  for (int i = 0; i < 4; ++i)
#pragma unroll
    for (int j = 0; j < 2; ++j) {
      const int nc = n0 + j * 16 + col_l;
      const int mr = m0 + i * 16 + half8;
      const float bv = bias[nc];
#pragma unroll
      for (int r = 0; r < 8; ++r) {
        const size_t off = (size_t)(mr + r) * (size_t)N + nc;
        float v = acc[i][j][r] + bv;
        if (MODE == 2) v += Rsd[off];
        if (MODE == 3) {
          v = 0.5f * v * (1.0f + erff(v * 0.70710678118654752f));
          Cb[off] = (__bf16)v;
        } else {
          Cf[off] = v;
        }
      }
    }
}

// ---------------------------------------------------------------------------
// Split fused QKV (f32, [B*T, 3C], bias already applied) into bf16
// Q,K: [B,H,T,64] row-major;  V: stored transposed Vt: [B,H,64,T]
// ---------------------------------------------------------------------------
__global__ __launch_bounds__(256) void qkv_split_kernel(
    const float* __restrict__ qkv, __bf16* __restrict__ Q,
    __bf16* __restrict__ Kb, __bf16* __restrict__ Vt) {
  const size_t idx = (size_t)blockIdx.x * 256 + threadIdx.x;  // < 4096*3072
  const int m = (int)(idx / 3072);
  const int c = (int)(idx % 3072);
  const int which = c >> 10;       // 0=q 1=k 2=v
  const int cc = c & 1023;
  const int h = cc >> 6;
  const int d = cc & 63;
  const int b = m >> 11;
  const int t = m & 2047;
  const __bf16 v = (__bf16)qkv[idx];
  const int bh = b * 16 + h;
  if (which == 0)      Q[((size_t)bh * 2048 + t) * 64 + d] = v;
  else if (which == 1) Kb[((size_t)bh * 2048 + t) * 64 + d] = v;
  else                 Vt[((size_t)bh * 64 + d) * 2048 + t] = v;
}

// ---------------------------------------------------------------------------
// Flash attention: 128 threads = 4 waves, each wave owns one 16-query tile of
// one (b,h). Streaming softmax; S = Q@K^T and O += P@V via WMMA. P is moved
// from C-layout to A-layout through a per-wave LDS staging tile.
// ctx output: bf16 [B*T, H*64] (heads re-interleaved for the O-projection).
// ---------------------------------------------------------------------------
__global__ __launch_bounds__(128) void attn_kernel(
    const __bf16* __restrict__ Q, const __bf16* __restrict__ Km,
    const __bf16* __restrict__ Vt, __bf16* __restrict__ ctx) {
  constexpr int T = 2048, DH = 64, CDIM = 1024;
  const int lane = threadIdx.x & 31;
  const int w = threadIdx.x >> 5;
  const int gw = blockIdx.x * 4 + w;   // 0..4095
  const int qt = gw & 127;             // q tile within sequence (T/16 = 128)
  const int bh = gw >> 7;              // 0..31  (= b*16 + h)
  const int h = bh & 15;
  const int b = bh >> 4;

  const __bf16* Qb = Q + (size_t)bh * T * DH;
  const __bf16* Kb = Km + (size_t)bh * T * DH;
  const __bf16* Vb = Vt + (size_t)bh * DH * T;

  __shared__ alignas(16) __bf16 lds_p[4][16 * 32];
  __bf16* lp = &lds_p[w][0];

  const int q0 = qt * 16;
  const bf16x16 aq0 = load_frag(Qb, DH, q0, 0, lane);
  const bf16x16 aq1 = load_frag(Qb, DH, q0, 32, lane);

  const f32x8 zero = {0.f, 0.f, 0.f, 0.f, 0.f, 0.f, 0.f, 0.f};
  f32x8 acc[4];
#pragma unroll
  for (int t = 0; t < 4; ++t) acc[t] = zero;
  float rm[8], rl[8];
#pragma unroll
  for (int r = 0; r < 8; ++r) { rm[r] = -1e30f; rl[r] = 0.f; }

  const float scale = 0.125f;  // 1/sqrt(64)
  const int half8 = (lane & 16) ? 8 : 0;
  const int col = lane & 15;

  for (int j0 = 0; j0 < T; j0 += 32) {
    // S tiles for keys [j0, j0+16) and [j0+16, j0+32), K-dim = 64 (2 chunks)
    f32x8 s0 = zero, s1 = zero;
    s0 = wmma_bf16(aq0, load_frag(Kb, DH, j0, 0, lane), s0);
    s0 = wmma_bf16(aq1, load_frag(Kb, DH, j0, 32, lane), s0);
    s1 = wmma_bf16(aq0, load_frag(Kb, DH, j0 + 16, 0, lane), s1);
    s1 = wmma_bf16(aq1, load_frag(Kb, DH, j0 + 16, 32, lane), s1);

#pragma unroll
    for (int r = 0; r < 8; ++r) {
      float v0 = s0[r] * scale, v1 = s1[r] * scale;
      float mx = fmaxf(v0, v1);
#pragma unroll
      for (int m = 8; m; m >>= 1) mx = fmaxf(mx, __shfl_xor(mx, m, 16));
      const float nm = fmaxf(rm[r], mx);
      const float sc = __expf(rm[r] - nm);
      rm[r] = nm;
      const float p0 = __expf(v0 - nm);
      const float p1 = __expf(v1 - nm);
      float rs = p0 + p1;
#pragma unroll
      for (int m = 8; m; m >>= 1) rs += __shfl_xor(rs, m, 16);
      rl[r] = rl[r] * sc + rs;
#pragma unroll
      for (int t = 0; t < 4; ++t) acc[t][r] *= sc;
      // stage P (row-major 16x32 bf16) for the A-layout reload
      lp[(r + half8) * 32 + col] = (__bf16)p0;
      lp[(r + half8) * 32 + col + 16] = (__bf16)p1;
    }
    __builtin_amdgcn_wave_barrier();
    const bf16x16 pf = load_frag(lp, 32, 0, 0, lane);
    // O(16x64) += P(16x32) @ V(32x64); Vt rows are d, contiguous in j
#pragma unroll
    for (int t = 0; t < 4; ++t)
      acc[t] = wmma_bf16(pf, load_frag(Vb, T, t * 16, j0, lane), acc[t]);
    __builtin_amdgcn_wave_barrier();
  }

  const size_t row0 = (size_t)b * T + q0 + half8;
#pragma unroll
  for (int t = 0; t < 4; ++t)
#pragma unroll
    for (int r = 0; r < 8; ++r) {
      const float o = acc[t][r] / rl[r];
      ctx[(row0 + r) * CDIM + h * 64 + t * 16 + col] = (__bf16)o;
    }
}

// ---------------------------------------------------------------------------
// Host orchestration
// ---------------------------------------------------------------------------
extern "C" void kernel_launch(void* const* d_in, const int* in_sizes, int n_in,
                              void* d_out, int out_size, void* d_ws,
                              size_t ws_size, hipStream_t stream) {
  constexpr int BT = 4096;   // B*T tokens
  constexpr int C = 1024;
  constexpr int TQ = 3072;   // 3*C
  constexpr int FF = 4096;

  const float* x      = (const float*)d_in[0];
  const float* qkv_w  = (const float*)d_in[1];
  const float* qkv_b  = (const float*)d_in[2];
  const float* o_w    = (const float*)d_in[3];
  const float* o_b    = (const float*)d_in[4];
  const float* ln1_g  = (const float*)d_in[5];
  const float* ln1_b  = (const float*)d_in[6];
  const float* ffn_w1 = (const float*)d_in[7];
  const float* ffn_b1 = (const float*)d_in[8];
  const float* ffn_w2 = (const float*)d_in[9];
  const float* ffn_b2 = (const float*)d_in[10];
  const float* ln2_g  = (const float*)d_in[11];
  const float* ln2_b  = (const float*)d_in[12];

  char* ws = (char*)d_ws;
  size_t off = 0;
  auto take = [&](size_t bytes) -> void* {
    void* p = ws + off;
    off += (bytes + 255) & ~(size_t)255;
    return p;
  };

  __bf16* qkvw_bf = (__bf16*)take((size_t)TQ * C * 2);
  __bf16* ow_bf   = (__bf16*)take((size_t)C * C * 2);
  __bf16* w1_bf   = (__bf16*)take((size_t)FF * C * 2);
  __bf16* w2_bf   = (__bf16*)take((size_t)C * FF * 2);
  __bf16* h_bf    = (__bf16*)take((size_t)BT * C * 2);      // LN1 out, reused for LN2 out
  float*  qkv_f32 = (float*)take((size_t)BT * TQ * 4);      // reused for ffn1 bf16
  __bf16* ffn1_bf = (__bf16*)qkv_f32;                       // 33.5MB <= 50MB
  __bf16* Qbf     = (__bf16*)take((size_t)BT * C * 2);
  __bf16* Kbf     = (__bf16*)take((size_t)BT * C * 2);
  __bf16* Vtbf    = (__bf16*)take((size_t)BT * C * 2);
  __bf16* ctx_bf  = (__bf16*)take((size_t)BT * C * 2);
  float*  x2      = (float*)take((size_t)BT * C * 4);
  float*  outp    = (float*)d_out;

  // 1) weight conversions fp32 -> bf16
  f32_to_bf16_kernel<<<(TQ * C + 255) / 256, 256, 0, stream>>>(qkv_w, qkvw_bf, TQ * C);
  f32_to_bf16_kernel<<<(C * C + 255) / 256, 256, 0, stream>>>(o_w, ow_bf, C * C);
  f32_to_bf16_kernel<<<(FF * C + 255) / 256, 256, 0, stream>>>(ffn_w1, w1_bf, FF * C);
  f32_to_bf16_kernel<<<(C * FF + 255) / 256, 256, 0, stream>>>(ffn_w2, w2_bf, C * FF);

  // 2) LN1
  ln_bf16_kernel<<<BT, 256, 0, stream>>>(x, ln1_g, ln1_b, h_bf, C);

  // 3) QKV = h @ qkv_w^T + b  (f32 out)
  gemm_bf16_kernel<0><<<dim3(TQ / 128, BT / 128), 256, 0, stream>>>(
      h_bf, qkvw_bf, qkv_b, nullptr, qkv_f32, nullptr, BT, TQ, C);

  // 4) split into Q,K [B,H,T,64] and Vt [B,H,64,T]
  qkv_split_kernel<<<(size_t)BT * TQ / 256, 256, 0, stream>>>(qkv_f32, Qbf, Kbf, Vtbf);

  // 5) flash attention -> ctx [B*T, C]
  attn_kernel<<<1024, 128, 0, stream>>>(Qbf, Kbf, Vtbf, ctx_bf);

  // 6) x2 = ctx @ o_w^T + o_b + x
  gemm_bf16_kernel<2><<<dim3(C / 128, BT / 128), 256, 0, stream>>>(
      ctx_bf, ow_bf, o_b, x, x2, nullptr, BT, C, C);

  // 7) LN2
  ln_bf16_kernel<<<BT, 256, 0, stream>>>(x2, ln2_g, ln2_b, h_bf, C);

  // 8) ffn1 = gelu(h2 @ w1^T + b1)  (bf16 out)
  gemm_bf16_kernel<3><<<dim3(FF / 128, BT / 128), 256, 0, stream>>>(
      h_bf, w1_bf, ffn_b1, nullptr, nullptr, ffn1_bf, BT, FF, C);

  // 9) out = ffn1 @ w2^T + b2 + x2
  gemm_bf16_kernel<2><<<dim3(C / 128, BT / 128), 256, 0, stream>>>(
      ffn1_bf, w2_bf, ffn_b2, x2, outp, nullptr, BT, C, FF);
}
